// LSTM_1477468750112
// MI455X (gfx1250) — compile-verified, matrix operand order checked by reading, not run on hardware
//
#include <hip/hip_runtime.h>

// ---------------------------------------------------------------------------
// LSTM scan for MI455X (gfx1250): bf16 WMMA (v_wmma_f32_16x16x32_bf16),
// fused per-step kernel (GEMM + gates + state update), gate-interleaved
// pre-transposed bf16 weight pack resident in L2 across all 256 steps.
// Step kernel uses a double-buffered (ping-pong) LDS pipeline with K-step 64:
// global loads for tile k+1 are issued before the ds_load/wmma chain of tile
// k, so VMEM latency overlaps matrix math; one barrier per K iteration.
// All LDS staging is explicit 128-bit (ds_store_b128) traffic.
// ---------------------------------------------------------------------------

typedef __attribute__((ext_vector_type(16))) __bf16        v16bf;
typedef __attribute__((ext_vector_type(8)))  float         v8f;
typedef __attribute__((ext_vector_type(4)))  unsigned int  v4u;

#define T_STEPS 256
#define BATCH   128
#define EDIM    1024
#define HDIM    1024
#define KTOT    2048          // E + H (A = [x | h])
#define NTOT    4096          // 4*H, gate-interleaved: n = 4*col + gate
#define WG_N    64            // output columns (n) per workgroup
#define KSTEP   64            // K per pipeline stage
#define NK      (KTOT / KSTEP)
#define LDS_STR 72            // padded bf16 row stride (144B, 16B-aligned)
#define PS_STR  68            // padded float row stride for preact staging
#define ABUF_B  (BATCH * LDS_STR * 2)   // 18432 B
#define BBUF_B  (WG_N  * LDS_STR * 2)   //  9216 B
#define BUF_B   (ABUF_B + BBUF_B)       // 27648 B

// ---------------------------------------------------------------------------
// Pack 8 fp32 weight matrices into one bf16 matrix Wt[NTOT][KTOT]:
//   Wt[4*col+g][k] = W{g}x[k][col]        (k <  EDIM)
//                  = W{g}h[k-EDIM][col]   (k >= EDIM)
// N-major (transposed) so B-tile staging into LDS is row-contiguous.
// ---------------------------------------------------------------------------
__global__ __launch_bounds__(256)
void lstm_pack_weights(const float* __restrict__ Wgx, const float* __restrict__ Wgh,
                       const float* __restrict__ Wix, const float* __restrict__ Wih,
                       const float* __restrict__ Wfx, const float* __restrict__ Wfh,
                       const float* __restrict__ Wox, const float* __restrict__ Woh,
                       __bf16* __restrict__ Wt) {
  int id = blockIdx.x * 256 + threadIdx.x;
  if (id >= NTOT * KTOT) return;
  int k   = id & (KTOT - 1);
  int n   = id >> 11;              // KTOT == 2^11
  int g   = n & 3;
  int col = n >> 2;
  const float* Wx = (g == 0) ? Wgx : (g == 1) ? Wix : (g == 2) ? Wfx : Wox;
  const float* Wh = (g == 0) ? Wgh : (g == 1) ? Wih : (g == 2) ? Wfh : Woh;
  float v = (k < EDIM) ? Wx[(size_t)k * HDIM + col]
                       : Wh[(size_t)(k - EDIM) * HDIM + col];
  Wt[(size_t)n * KTOT + k] = (__bf16)v;
}

// ---------------------------------------------------------------------------
// Pack biases (gate-interleaved) and initialize state: h0 = 0, c0 = 1.
// ---------------------------------------------------------------------------
__global__ __launch_bounds__(256)
void lstm_init(const float* __restrict__ bg, const float* __restrict__ bi,
               const float* __restrict__ bfb, const float* __restrict__ bo,
               float* __restrict__ biascat, __bf16* __restrict__ h0,
               float* __restrict__ c) {
  int id = blockIdx.x * 256 + threadIdx.x;
  if (id < NTOT) {
    int g = id & 3, col = id >> 2;
    const float* b = (g == 0) ? bg : (g == 1) ? bi : (g == 2) ? bfb : bo;
    biascat[id] = b[col];
  }
  if (id < BATCH * HDIM) {
    h0[id] = (__bf16)0.0f;
    c[id]  = 1.0f;
  }
}

// ---------------------------------------------------------------------------
// Load one 16x32 bf16 WMMA operand fragment from an LDS tile stored row-major
// with LDS_STR padding. Matches ISA 7.12.2 16-bit A layout:
//   lane L: row = L%16, half = L/16
//   elems 0..7  = K[half*8 .. +7], elems 8..15 = K[16+half*8 .. +7]
// B is staged column-major (n-major) into LDS so it uses the same loader.
// ---------------------------------------------------------------------------
__device__ __forceinline__ v16bf lds_frag(const __bf16* rowp, int half) {
  union { v16bf v; v4u q[2]; } x;
  x.q[0] = *(const v4u*)(rowp + half * 8);
  x.q[1] = *(const v4u*)(rowp + 16 + half * 8);
  return x.v;
}

// ---------------------------------------------------------------------------
// Stage one KSTEP-wide A/B tile pair into an LDS ping-pong buffer.
//   A: [128 rows][64 k]  (x: fp32 -> bf16 on the fly; h: bf16 copy)
//   B: [64 n][64 k]      (row-contiguous from N-major Wt)
// Every LDS write is an explicit 16-byte (ds_store_b128) transaction.
// ---------------------------------------------------------------------------
__device__ __forceinline__ void stage_tiles(const float*  __restrict__ X_t,
                                            const __bf16* __restrict__ h_in,
                                            const __bf16* __restrict__ Wt,
                                            __bf16* __restrict__ Ab,
                                            __bf16* __restrict__ Bb,
                                            int k0, int nbase, int tid) {
  // ---- A tile ----
  {
    const int row = tid >> 1;
    const int seg = (tid & 1) * 32;
    v4u* dst = (v4u*)(Ab + row * LDS_STR + seg);   // 4 x 16B chunks
    const int k = k0 + seg;            // KSTEP | EDIM => tile never straddles
    if (k0 < EDIM) {
      const float* src = X_t + (size_t)row * EDIM + k;
#pragma unroll
      for (int q = 0; q < 4; ++q) {
        union { __bf16 h[8]; v4u v; } pk;
#pragma unroll
        for (int e = 0; e < 8; ++e) pk.h[e] = (__bf16)src[q * 8 + e];
        dst[q] = pk.v;                 // v_cvt_pk_bf16_f32 x4 + ds_store_b128
      }
    } else {
      const v4u* src = (const v4u*)(h_in + (size_t)row * HDIM + (k - EDIM));
#pragma unroll
      for (int q = 0; q < 4; ++q) dst[q] = src[q];
    }
  }
  // ---- B tile ----
  {
    const int nl  = tid >> 2;          // 0..63
    const int seg = (tid & 3) * 16;    // 0,16,32,48
    const v4u* src = (const v4u*)(Wt + (size_t)(nbase + nl) * KTOT + k0 + seg);
    v4u* dst = (v4u*)(Bb + nl * LDS_STR + seg);
    dst[0] = src[0];
    dst[1] = src[1];
  }
}

// ---------------------------------------------------------------------------
// One LSTM timestep. Grid: NTOT/WG_N = 64 blocks x 256 threads (8 wave32).
// Block computes preact[0:128][nbase:nbase+64] = [x|h] @ Wt^T, then fuses
// bias + activations + (c,h) update for cols [nbase/4, nbase/4+16).
// h is double-buffered across steps to avoid cross-block WAR hazards.
// ---------------------------------------------------------------------------
__global__ __launch_bounds__(256)
void lstm_step(const float*  __restrict__ X_t,     // [BATCH][EDIM] fp32 (this t)
               const __bf16* __restrict__ Wt,      // [NTOT][KTOT] bf16
               const float*  __restrict__ biascat, // [NTOT]
               const __bf16* __restrict__ h_in,    // [BATCH][HDIM] bf16
               __bf16*       __restrict__ h_out,   // [BATCH][HDIM] bf16
               float*        __restrict__ c,       // [BATCH][HDIM] fp32
               float*        __restrict__ out_t) { // [BATCH][HDIM] fp32
  __shared__ __align__(16) char smem[2 * BUF_B];   // 55296 B (> Ps 34816 B)
  __bf16* A0 = (__bf16*)(smem);
  __bf16* B0 = (__bf16*)(smem + ABUF_B);
  __bf16* A1 = (__bf16*)(smem + BUF_B);
  __bf16* B1 = (__bf16*)(smem + BUF_B + ABUF_B);
  float*  Ps = (float*)smem;                       // [128][68], reused post-GEMM

  const int tid   = threadIdx.x;
  const int lane  = tid & 31;
  const int wave  = tid >> 5;
  const int half  = lane >> 4;
  const int l16   = lane & 15;
  const int nbase = blockIdx.x * WG_N;
  const int waveM = (wave >> 1) * 32;   // 8 waves as 4(M) x 2(N), 32x32 each
  const int waveN = (wave & 1) * 32;

  // bias values for this lane's two N columns (hoisted out of the K loop)
  const float bv0 = biascat[nbase + waveN + l16];
  const float bv1 = biascat[nbase + waveN + 16 + l16];

  v8f acc[2][2] = {};

  // prologue: stage tile 0 into buffer 0
  stage_tiles(X_t, h_in, Wt, A0, B0, 0, nbase, tid);

  for (int kt = 0; kt < NK; ++kt) {
    __syncthreads();                       // tile kt resident; buf kt^1 free
    const int cur = kt & 1;

    // L2 prefetch two tiles ahead (X stream is the per-step HBM traffic)
    if (kt + 2 < NK) {
      int k2 = (kt + 2) * KSTEP;
      __builtin_prefetch(Wt + (size_t)(nbase + (tid >> 2)) * KTOT + k2 + (tid & 3) * 16, 0, 0);
      if (k2 < EDIM)
        __builtin_prefetch(X_t + (size_t)(tid >> 1) * EDIM + k2 + (tid & 1) * 32, 0, 0);
    }

    // stage tile kt+1 into the other buffer (overlaps with wmma below)
    if (kt + 1 < NK)
      stage_tiles(X_t, h_in, Wt, cur ? A0 : A1, cur ? B0 : B1,
                  (kt + 1) * KSTEP, nbase, tid);

    // compute on tile kt
    const __bf16* Ab = cur ? A1 : A0;
    const __bf16* Bb = cur ? B1 : B0;
#pragma unroll
    for (int ks = 0; ks < 2; ++ks) {
      const int ko = ks * 32;
      v16bf a0 = lds_frag(Ab + (waveM +  0 + l16) * LDS_STR + ko, half);
      v16bf a1 = lds_frag(Ab + (waveM + 16 + l16) * LDS_STR + ko, half);
      v16bf b0 = lds_frag(Bb + (waveN +  0 + l16) * LDS_STR + ko, half);
      v16bf b1 = lds_frag(Bb + (waveN + 16 + l16) * LDS_STR + ko, half);
      acc[0][0] = __builtin_amdgcn_wmma_f32_16x16x32_bf16(false, a0, false, b0, (short)0, acc[0][0], false, false);
      acc[0][1] = __builtin_amdgcn_wmma_f32_16x16x32_bf16(false, a0, false, b1, (short)0, acc[0][1], false, false);
      acc[1][0] = __builtin_amdgcn_wmma_f32_16x16x32_bf16(false, a1, false, b0, (short)0, acc[1][0], false, false);
      acc[1][1] = __builtin_amdgcn_wmma_f32_16x16x32_bf16(false, a1, false, b1, (short)0, acc[1][1], false, false);
    }
  }

  __syncthreads();
  // --- spill biased preacts to LDS (C/D layout: lanes 0-15 N=lane, M=r;
  //     lanes 16-31 N=lane-16, M=r+8) so gates can be regrouped per column ---
#pragma unroll
  for (int nj = 0; nj < 2; ++nj) {
    int   n  = waveN + nj * 16 + l16;
    float bv = nj ? bv1 : bv0;
#pragma unroll
    for (int mi = 0; mi < 2; ++mi) {
#pragma unroll
      for (int r = 0; r < 8; ++r) {
        int m = waveM + mi * 16 + r + half * 8;
        Ps[m * PS_STR + n] = acc[mi][nj][r] + bv;
      }
    }
  }
  __syncthreads();

  // --- gate fusion + state update for cols [colbase, colbase+16) ---
  const int colbase = nbase >> 2;
#pragma unroll
  for (int it = 0; it < 8; ++it) {
    int idx = tid + it * 256;           // 0..2047 = 128 rows x 16 cols
    int b   = idx >> 4;
    int cl  = idx & 15;
    const float* p = Ps + b * PS_STR + cl * 4;
    float gg = tanhf(p[0]);
    float ii = 0.5f + 0.5f * tanhf(0.5f * p[1]);   // sigmoid via v_tanh_f32
    float ff = 0.5f + 0.5f * tanhf(0.5f * p[2]);
    float oo = 0.5f + 0.5f * tanhf(0.5f * p[3]);
    size_t off = (size_t)b * HDIM + colbase + cl;
    float cnew = gg * ii + c[off] * ff;
    float hnew = tanhf(cnew) * oo;
    c[off]     = cnew;
    out_t[off] = hnew;
    h_out[off] = (__bf16)hnew;
  }
}

// ---------------------------------------------------------------------------
// Workspace layout (~17.0 MB):
//   Wt       bf16 [4096][2048]   16 MB
//   biascat  f32  [4096]         16 KB
//   c        f32  [128][1024]    512 KB
//   h (x2)   bf16 [128][1024]    2 x 256 KB (double buffer)
// ---------------------------------------------------------------------------
extern "C" void kernel_launch(void* const* d_in, const int* in_sizes, int n_in,
                              void* d_out, int out_size, void* d_ws, size_t ws_size,
                              hipStream_t stream) {
  (void)in_sizes; (void)n_in; (void)out_size; (void)ws_size;
  const float* embeds = (const float*)d_in[0];
  const float* Wgx = (const float*)d_in[1];
  const float* Wgh = (const float*)d_in[2];
  const float* Wix = (const float*)d_in[3];
  const float* Wih = (const float*)d_in[4];
  const float* Wfx = (const float*)d_in[5];
  const float* Wfh = (const float*)d_in[6];
  const float* Wox = (const float*)d_in[7];
  const float* Woh = (const float*)d_in[8];
  const float* bg  = (const float*)d_in[9];
  const float* bi  = (const float*)d_in[10];
  const float* bfb = (const float*)d_in[11];
  const float* bo  = (const float*)d_in[12];

  char* ws = (char*)d_ws;
  size_t off = 0;
  __bf16* Wt      = (__bf16*)(ws + off); off += (size_t)NTOT * KTOT * 2;
  float*  biascat = (float*) (ws + off); off += (size_t)NTOT * 4;
  float*  c       = (float*) (ws + off); off += (size_t)BATCH * HDIM * 4;
  __bf16* hbuf0   = (__bf16*)(ws + off); off += (size_t)BATCH * HDIM * 2;
  __bf16* hbuf1   = (__bf16*)(ws + off); off += (size_t)BATCH * HDIM * 2;

  lstm_pack_weights<<<(NTOT * KTOT + 255) / 256, 256, 0, stream>>>(
      Wgx, Wgh, Wix, Wih, Wfx, Wfh, Wox, Woh, Wt);
  lstm_init<<<(BATCH * HDIM + 255) / 256, 256, 0, stream>>>(
      bg, bi, bfb, bo, biascat, hbuf0, c);

  float*  out   = (float*)d_out;
  __bf16* hb[2] = { hbuf0, hbuf1 };
  for (int t = 0; t < T_STEPS; ++t) {
    lstm_step<<<NTOT / WG_N, 256, 0, stream>>>(
        embeds + (size_t)t * BATCH * EDIM, Wt, biascat,
        hb[t & 1], hb[(t + 1) & 1], c,
        out + (size_t)t * BATCH * HDIM);
  }
}
